// GraphAttentionLayer_54331336295085
// MI455X (gfx1250) — matrix operand order repaired and checked by reading, not compiled, not run on hardware
//
#include <hip/hip_runtime.h>
#include <hip/hip_bf16.h>
#include <math.h>

typedef __attribute__((ext_vector_type(16))) _Float16 v16h;
typedef __attribute__((ext_vector_type(8)))  _Float16 v8h;
typedef __attribute__((ext_vector_type(8)))  float    v8f;

#define Nn 8192
#define Ff 256
#define Dd 64
#define Kk 2
#define NW (Nn / 32)   // 32-bit mask words per adjacency row

// ---------------------------------------------------------------- fp32 -> f16
__global__ void cvt_f16_kernel(const float* __restrict__ src,
                               _Float16* __restrict__ dst, int n) {
    int t = blockIdx.x * blockDim.x + threadIdx.x;
    if (t < n) dst[t] = (_Float16)src[t];
}

// ---------------------------------------- W [K,F,D] fp32 -> WT16 [K,D,F] f16
__global__ void cvt_w_kernel(const float* __restrict__ W,
                             _Float16* __restrict__ WT16) {
    int t = blockIdx.x * blockDim.x + threadIdx.x;
    if (t >= Kk * Ff * Dd) return;
    int k = t / (Ff * Dd);
    int f = (t / Dd) % Ff;
    int d = t % Dd;
    WT16[((size_t)k * Dd + d) * Ff + f] = (_Float16)W[t];
}

// ------------------------------------------------- pack A (int32) -> bitmask
__global__ void pack_mask_kernel(const int* __restrict__ A,
                                 unsigned* __restrict__ bits) {
    size_t t = (size_t)blockIdx.x * blockDim.x + threadIdx.x;
    unsigned long long bal = __ballot(A[t] != 0);
    if ((threadIdx.x & 31) == 0) bits[t >> 5] = (unsigned)bal;
}

// ------------------------------- HW^T = (H x W)^T  (WMMA f16) -> HWT16[k][d][n]
__global__ void hw_gemm_kernel(const _Float16* __restrict__ H16,
                               const _Float16* __restrict__ WT16,
                               _Float16* __restrict__ HWT16) {
    const int lane = threadIdx.x & 31;
    const int wid  = (blockIdx.x * blockDim.x + threadIdx.x) >> 5;
    const int k    = wid / (Nn / 16);
    const int i0   = (wid % (Nn / 16)) * 16;
    const int row  = lane & 15;            // A-matrix M
    const int hi8  = (lane >> 4) * 8;      // A-matrix K group offset
    const int col  = lane & 15;            // B-matrix N
    const int kb   = (lane >> 4) * 8;      // B-matrix K group offset

    v8f c[4] = {};
    for (int f0 = 0; f0 < Ff; f0 += 32) {
        const _Float16* ap = H16 + (size_t)(i0 + row) * Ff + f0 + hi8;
        v8h alo = *(const v8h*)ap;
        v8h ahi = *(const v8h*)(ap + 16);
        v16h a = __builtin_shufflevector(alo, ahi, 0, 1, 2, 3, 4, 5, 6, 7,
                                         8, 9, 10, 11, 12, 13, 14, 15);
#pragma unroll
        for (int g = 0; g < 4; ++g) {
            const _Float16* bp =
                WT16 + ((size_t)k * Dd + g * 16 + col) * Ff + f0 + kb;
            v8h blo = *(const v8h*)bp;
            v8h bhi = *(const v8h*)(bp + 16);
            v16h b = __builtin_shufflevector(blo, bhi, 0, 1, 2, 3, 4, 5, 6, 7,
                                             8, 9, 10, 11, 12, 13, 14, 15);
            c[g] = __builtin_amdgcn_wmma_f32_16x16x32_f16(
                false, a, false, b, (short)0, c[g], false, false);
        }
    }
    const int orow = (lane >> 4) * 8;      // C/D layout: lanes 16-31 hold M=8..15
#pragma unroll
    for (int g = 0; g < 4; ++g)
#pragma unroll
        for (int r = 0; r < 8; ++r)
            HWT16[((size_t)k * Dd + g * 16 + col) * Nn + i0 + orow + r] =
                (_Float16)c[g][r];
}

// ---------------------------------------------- e1/e2 = HW . a1 / HW . a2
// reads HWT16[k][d][n]: for fixed d, consecutive threads read consecutive n.
__global__ void e_proj_kernel(const _Float16* __restrict__ HWT16,
                              const float* __restrict__ a1,
                              const float* __restrict__ a2,
                              float* __restrict__ e1, float* __restrict__ e2) {
    int t = blockIdx.x * blockDim.x + threadIdx.x;
    if (t >= Kk * Nn) return;
    int k = t / Nn, n = t % Nn;
    const _Float16* hp = HWT16 + (size_t)k * Dd * Nn + n;
    float s1 = 0.f, s2 = 0.f;
    for (int d = 0; d < Dd; ++d) {
        float h = (float)hp[(size_t)d * Nn];
        s1 += h * a1[k * Dd + d];
        s2 += h * a2[k * Dd + d];
    }
    e1[t] = s1;
    e2[t] = s2;
}

// ------------------------------- row softmax stats: m[k,j], invZ[k,j]
__global__ void softmax_stats_kernel(const float* __restrict__ e1,
                                     const float* __restrict__ e2,
                                     const unsigned* __restrict__ bits,
                                     float* __restrict__ mrow,
                                     float* __restrict__ izrow) {
    const int lane = threadIdx.x & 31;
    const int wid  = (blockIdx.x * blockDim.x + threadIdx.x) >> 5;
    const int k = wid / Nn;
    const int j = wid % Nn;
    const float ej = e1[k * Nn + j];
    float m = -3.0e38f, z = 0.f;
    for (int it = 0; it < NW; ++it) {
        const unsigned w = bits[(size_t)j * NW + it];   // wave-uniform word
        if ((w >> lane) & 1u) {
            float x = ej + e2[k * Nn + it * 32 + lane];
            x = (x >= 0.f) ? x : 0.2f * x;              // LeakyReLU(0.2)
            if (x > m) { z *= __expf(m - x); m = x; }
            z += __expf(x - m);
        }
    }
#pragma unroll
    for (int off = 16; off; off >>= 1) {                // wave32 online-merge
        float om = __shfl_xor(m, off, 32);
        float oz = __shfl_xor(z, off, 32);
        float nm = fmaxf(m, om);
        z = z * __expf(m - nm) + oz * __expf(om - nm);
        m = nm;
    }
    if (lane == 0) {
        mrow[k * Nn + j]  = m;
        izrow[k * Nn + j] = 1.0f / z;
    }
}

// -------------------- out = attn^T @ HW + b, ReLU  (fused, WMMA f16)
// block = 4 waves, one k, four adjacent 16-row i-tiles.
// HWT j-tile [64 d][32 j] staged into LDS via global_load_async_to_lds_b128.
__global__ void attn_gemm_kernel(const _Float16* __restrict__ HWT16,
                                 const float* __restrict__ e1,
                                 const float* __restrict__ e2,
                                 const float* __restrict__ mrow,
                                 const float* __restrict__ izrow,
                                 const unsigned* __restrict__ bits,
                                 const float* __restrict__ bvec,
                                 float* __restrict__ out) {
    __shared__ _Float16 lbT[Dd][40];   // 64 d-rows x 32 j (+pad); 80B rows, 16B-aligned
    const int lane = threadIdx.x & 31;
    const int wv   = threadIdx.x >> 5;                      // 0..3
    const int k    = blockIdx.x / (Nn / 64);
    const int i0   = (blockIdx.x % (Nn / 64)) * 64 + wv * 16;
    const int row  = lane & 15;
    const int hi8  = (lane >> 4) * 8;
    const int col  = lane & 15;
    const int kb   = (lane >> 4) * 8;
    const int wsel = i0 >> 5;               // mask word holding this i-tile
    const int bsh  = i0 & 31;               // 0 or 16
    const float my_e2 = e2[k * Nn + i0 + row];

    // per-thread staging addresses (constant across j-loop except global ptr)
    const int sr = threadIdx.x >> 1;        // 0..63  (d row)
    const int ss = (threadIdx.x & 1) * 16;  // 0/16   (j halfs)
    const unsigned lds_a =
        (unsigned)(unsigned long long)(const void*)&lbT[sr][ss];

    v8f c[4] = {};
    for (int j0 = 0; j0 < Nn; j0 += 32) {
        __syncthreads();                    // previous iteration's reads done
        {   // async DMA: HWT16[k][sr][j0+ss .. +15] -> LDS (2 x 16B beats)
            const _Float16* src =
                HWT16 + ((size_t)k * Dd + sr) * Nn + j0 + ss;
            unsigned long long ga = (unsigned long long)src;
            asm volatile("global_load_async_to_lds_b128 %0, %1, off"
                         :: "v"(lds_a), "v"(ga) : "memory");
            asm volatile("global_load_async_to_lds_b128 %0, %1, off offset:16"
                         :: "v"(lds_a), "v"(ga) : "memory");
            asm volatile("s_wait_asynccnt 0x0" ::: "memory");
        }
        __syncthreads();                    // all LDS writes visible

        // per-lane stats for j = j0+lane, redistributed via shfl
        const int jj = j0 + lane;
        const float    wj_e1 = e1[k * Nn + jj];
        const float    wj_m  = mrow[k * Nn + jj];
        const float    wj_iz = izrow[k * Nn + jj];
        const unsigned wj_b  = bits[(size_t)jj * NW + wsel];

        v16h a;
#pragma unroll
        for (int e = 0; e < 16; ++e) {
            const int jrel = (e < 8) ? (hi8 + e) : (16 + hi8 + (e - 8));
            const float    je1 = __shfl(wj_e1, jrel, 32);
            const float    jm  = __shfl(wj_m,  jrel, 32);
            const float    jiz = __shfl(wj_iz, jrel, 32);
            const unsigned jb  = (unsigned)__shfl((int)wj_b, jrel, 32);
            float x = je1 + my_e2;
            x = (x >= 0.f) ? x : 0.2f * x;
            float p = ((jb >> (bsh + row)) & 1u) ? __expf(x - jm) * jiz : 0.0f;
            a[e] = (_Float16)p;
        }
#pragma unroll
        for (int g = 0; g < 4; ++g) {
            // B operand: two 8-half runs per lane (k=kb..+7, 16+kb..+7) -> 2x ds_load_b128
            const _Float16* bp = &lbT[g * 16 + col][kb];
            v8h blo = *(const v8h*)bp;
            v8h bhi = *(const v8h*)(bp + 16);
            v16h b = __builtin_shufflevector(blo, bhi, 0, 1, 2, 3, 4, 5, 6, 7,
                                             8, 9, 10, 11, 12, 13, 14, 15);
            c[g] = __builtin_amdgcn_wmma_f32_16x16x32_f16(
                false, a, false, b, (short)0, c[g], false, false);
        }
    }

    const int orow = (lane >> 4) * 8;
#pragma unroll
    for (int g = 0; g < 4; ++g) {
        const float bb = bvec[k * Dd + g * 16 + col];
#pragma unroll
        for (int r = 0; r < 8; ++r) {
            float v = c[g][r] + bb;
            out[(size_t)(i0 + orow + r) * (Kk * Dd) + k * Dd + g * 16 + col] =
                v > 0.f ? v : 0.f;
        }
    }
}

// ---------------------------------------------------------------------------
extern "C" void kernel_launch(void* const* d_in, const int* in_sizes, int n_in,
                              void* d_out, int out_size, void* d_ws, size_t ws_size,
                              hipStream_t stream) {
    const float* H  = (const float*)d_in[0];
    const int*   A  = (const int*)d_in[1];
    const float* W  = (const float*)d_in[2];
    const float* b  = (const float*)d_in[3];
    const float* a1 = (const float*)d_in[4];
    const float* a2 = (const float*)d_in[5];
    float* out = (float*)d_out;

    char* ws = (char*)d_ws;
    _Float16* H16   = (_Float16*)ws;  ws += (size_t)Nn * Ff * 2;
    _Float16* WT16  = (_Float16*)ws;  ws += (size_t)Kk * Ff * Dd * 2;
    _Float16* HWT16 = (_Float16*)ws;  ws += (size_t)Kk * Nn * Dd * 2;
    float*    e1    = (float*)ws;     ws += (size_t)Kk * Nn * 4;
    float*    e2    = (float*)ws;     ws += (size_t)Kk * Nn * 4;
    float*    mrow  = (float*)ws;     ws += (size_t)Kk * Nn * 4;
    float*    iz    = (float*)ws;     ws += (size_t)Kk * Nn * 4;
    unsigned* bits  = (unsigned*)ws;  ws += (size_t)Nn * (Nn / 8);

    cvt_f16_kernel<<<(Nn * Ff + 255) / 256, 256, 0, stream>>>(H, H16, Nn * Ff);
    cvt_w_kernel<<<(Kk * Ff * Dd + 255) / 256, 256, 0, stream>>>(W, WT16);
    pack_mask_kernel<<<(int)(((size_t)Nn * Nn) / 256), 256, 0, stream>>>(A, bits);
    hw_gemm_kernel<<<Kk * (Nn / 16) / 4, 128, 0, stream>>>(H16, WT16, HWT16);
    e_proj_kernel<<<(Kk * Nn + 255) / 256, 256, 0, stream>>>(HWT16, a1, a2, e1, e2);
    softmax_stats_kernel<<<Kk * Nn / 4, 128, 0, stream>>>(e1, e2, bits, mrow, iz);
    attn_gemm_kernel<<<Kk * (Nn / 64), 128, 0, stream>>>(HWT16, e1, e2, mrow, iz,
                                                         bits, b, out);
}
